// Upsample_34660386078901
// MI455X (gfx1250) — matrix operand order repaired and checked by reading, not compiled, not run on hardware
//
#include <hip/hip_runtime.h>
#include <hip/hip_bf16.h>

#define BATCH 2
#define CH    64
#define NEXP  4
#define HLR   192
#define WLR   192
#define SCALE 2
#define HHR   (HLR*SCALE)   // 384
#define WHR   (WLR*SCALE)   // 384
#define NPIX  (HHR*WHR)     // 147456

typedef __attribute__((ext_vector_type(16))) _Float16 v16h;
typedef __attribute__((ext_vector_type(8)))  float    v8f;

static __device__ __forceinline__ v8f wmma_f16(v16h a, v16h b, v8f c) {
  // D = A(16x32 f16) * B(32x16 f16) + C(16x16 f32)
  return __builtin_amdgcn_wmma_f32_16x16x32_f16(
      /*neg_a=*/false, a, /*neg_b=*/false, b,
      /*c_mod=*/(short)0, c, /*reuse_a=*/false, /*reuse_b=*/false);
}

// ---------------------------------------------------------------------------
// Kernel 1: pack weight_compress [E,8,64] / weight_expand [E,64,8] into f16
// WMMA B-matrix layout in workspace. 8 B-matrices of 32x16:
//   mats 0..3: compress  (k_half, n_tile): B[k][n], n=e*8+o, val=Wc[e][o][k]
//   mats 4..7: expand    (c_tile):         B[kk][c], kk=e*8+o, val=We[e][c][o]
// B-layout per lane L: col N = L%16; halfs h -> K = 8*(L/16) + (h<8 ? h : 8+h)
// ---------------------------------------------------------------------------
__global__ void pack_weights_kernel(const float* __restrict__ wc,
                                    const float* __restrict__ we,
                                    _Float16* __restrict__ wsw) {
  int lane = threadIdx.x;
  if (lane >= 32) return;
  int nn    = lane & 15;
  int kbase = (lane < 16) ? 0 : 8;
  #pragma unroll
  for (int kh = 0; kh < 2; kh++) {
    #pragma unroll
    for (int nt = 0; nt < 2; nt++) {
      int mat = kh*2 + nt;
      #pragma unroll
      for (int h = 0; h < 16; h++) {
        int k = kh*32 + kbase + h + (h < 8 ? 0 : 8);  // channel index 0..63
        int n = nt*16 + nn;                  // 0..31 = e*8+o
        int e = n >> 3, o = n & 7;
        wsw[mat*512 + lane*16 + h] = (_Float16)wc[(e*8 + o)*64 + k];
      }
    }
  }
  #pragma unroll
  for (int ct = 0; ct < 4; ct++) {
    int mat = 4 + ct;
    #pragma unroll
    for (int h = 0; h < 16; h++) {
      int kk = kbase + h + (h < 8 ? 0 : 8);  // 0..31 = e*8+o
      int c  = ct*16 + nn;                   // output channel
      int e = kk >> 3, o = kk & 7;
      wsw[mat*512 + lane*16 + h] = (_Float16)we[(e*64 + c)*8 + o];
    }
  }
}

// ---------------------------------------------------------------------------
// Kernel 2: per-HR-pixel coordinate MLP -> sample coords (float2) and
// sigmoid routing weights (float4), stored to workspace. Batch-independent.
// ---------------------------------------------------------------------------
__global__ __launch_bounds__(256) void mlp_kernel(
    const float* __restrict__ w1, const float* __restrict__ b1,
    const float* __restrict__ w2, const float* __restrict__ b2,
    const float* __restrict__ rw, const float* __restrict__ rb,
    const float* __restrict__ ow, const float* __restrict__ ob,
    float2* __restrict__ coords, float4* __restrict__ routg) {
  __shared__ float s_w2[64*64];
  __shared__ float s_w1[256], s_b1[64], s_b2[64], s_rw[256], s_ow[128], s_misc[8];
  int t = threadIdx.x;
  for (int i = t; i < 4096; i += 256) s_w2[i] = w2[i];
  if (t < 256) s_w1[t] = w1[t];
  if (t < 64)  { s_b1[t] = b1[t]; s_b2[t] = b2[t]; }
  if (t < 256) s_rw[t] = rw[t];
  if (t < 128) s_ow[t] = ow[t];
  if (t < 4)   s_misc[t] = rb[t];
  if (t < 2)   s_misc[4 + t] = ob[t];
  __syncthreads();

  int p = blockIdx.x*256 + t;
  int y = p / WHR, x = p % WHR;
  const float inv = 1.0f / (float)SCALE;
  float chv = ((float)y + 0.5f) * inv;
  float cwv = ((float)x + 0.5f) * inv;
  float coorh = chv - floorf(chv + 0.001f) - 0.5f;
  float coorw = cwv - floorf(cwv + 0.001f) - 0.5f;

  float e1[64];
  #pragma unroll
  for (int o = 0; o < 64; o++) {
    float a = s_b1[o] + s_w1[o*4+0]*inv + s_w1[o*4+1]*inv
                      + s_w1[o*4+2]*coorh + s_w1[o*4+3]*coorw;
    e1[o] = fmaxf(a, 0.0f);
  }
  float offx = s_misc[4], offy = s_misc[5];
  float r0 = s_misc[0], r1 = s_misc[1], r2 = s_misc[2], r3 = s_misc[3];
  for (int o = 0; o < 64; o++) {
    float a = s_b2[o];
    #pragma unroll
    for (int i = 0; i < 64; i++) a += s_w2[o*64 + i] * e1[i];
    float e2 = fmaxf(a, 0.0f);
    offx += s_ow[o]      * e2;
    offy += s_ow[64 + o] * e2;
    r0 += s_rw[o]       * e2;
    r1 += s_rw[64 + o]  * e2;
    r2 += s_rw[128 + o] * e2;
    r3 += s_rw[192 + o] * e2;
  }
  float2 cd;
  cd.x = cwv - 0.5f + offx;   // absolute LR x coord
  cd.y = chv - 0.5f + offy;   // absolute LR y coord
  coords[p] = cd;
  float4 rr;
  rr.x = 1.0f / (1.0f + __expf(-r0));
  rr.y = 1.0f / (1.0f + __expf(-r1));
  rr.z = 1.0f / (1.0f + __expf(-r2));
  rr.w = 1.0f / (1.0f + __expf(-r3));
  routg[p] = rr;
}

// ---------------------------------------------------------------------------
// Kernel 3: main. One wave = one tile of 16 HR pixels (flat, row-aligned
// since 16 | WHR). Bilinear gather -> LDS, WMMA compress (K=64, N=32),
// routing mix + rescale, WMMA expand (K=32, N=64), +fea0, transposed
// contiguous stores. Both batches per wave. 4 waves / block.
// ---------------------------------------------------------------------------
__global__ __launch_bounds__(128) void upsample_main_kernel(
    const float*   __restrict__ fused,
    const _Float16* __restrict__ wsw,
    const float2*  __restrict__ coords,
    const float4*  __restrict__ routg,
    float*         __restrict__ out) {
  __shared__ __align__(16) float s_fea[4][16][68];   // [wave][pixel][channel]
  __shared__              float s_mid[4][16][36];    // [wave][pixel][e*8+o]
  __shared__ __align__(16) float s_out[4][64][16];   // [wave][channel][pixel]

  const int wave = threadIdx.x >> 5;
  const int lane = threadIdx.x & 31;
  const int lm   = lane & 15;        // pixel-in-tile (A/C row) / column idx
  const int hs   = lane >> 4;        // lane half
  const int tile = blockIdx.x*4 + wave;
  const int pb   = tile * 16;        // base flat HR pixel

  // Constant WMMA B matrices (packed by kernel 1), 32 B per lane per matrix.
  v16h Bc0 = *(const v16h*)(wsw + 0*512 + lane*16);
  v16h Bc1 = *(const v16h*)(wsw + 1*512 + lane*16);
  v16h Bc2 = *(const v16h*)(wsw + 2*512 + lane*16);
  v16h Bc3 = *(const v16h*)(wsw + 3*512 + lane*16);
  v16h Be0 = *(const v16h*)(wsw + 4*512 + lane*16);
  v16h Be1 = *(const v16h*)(wsw + 5*512 + lane*16);
  v16h Be2 = *(const v16h*)(wsw + 6*512 + lane*16);
  v16h Be3 = *(const v16h*)(wsw + 7*512 + lane*16);

  // Per-pixel sampling setup (pixel pb+lm, duplicated in both lane halves).
  float2 crd = coords[pb + lm];
  float4 rt  = routg[pb + lm];
  float rlo = hs ? rt.y : rt.x;      // routing weight for K-group of this half
  float rhi = hs ? rt.w : rt.z;

  float px = crd.x, py = crd.y;
  float fx = floorf(px), fy = floorf(py);
  float wx = px - fx, wy = py - fy;
  int xi = (int)fx, yi = (int)fy;
  float wgt[4]; int off[4];
  #pragma unroll
  for (int k = 0; k < 4; k++) {
    int dx = k & 1, dy = k >> 1;
    int xx = xi + dx, yy = yi + dy;
    bool valid = (xx >= 0) & (xx < WLR) & (yy >= 0) & (yy < HLR);
    float wv = (dx ? wx : 1.0f - wx) * (dy ? wy : 1.0f - wy);
    wgt[k] = valid ? wv : 0.0f;
    int xc = min(max(xx, 0), WLR - 1);
    int yc = min(max(yy, 0), HLR - 1);
    off[k] = yc*WLR + xc;
  }

  for (int b = 0; b < BATCH; b++) {
    const float* fb = fused + (size_t)b * CH * HLR * WLR;
    __syncthreads();
    // Bilinear gather: lane handles its pixel, 32 channels per lane half.
    #pragma unroll 4
    for (int c = 0; c < 32; c++) {
      int ca = hs*32 + c;
      const float* fc = fb + (size_t)ca * (HLR*WLR);
      float v = wgt[0]*fc[off[0]] + wgt[1]*fc[off[1]]
              + wgt[2]*fc[off[2]] + wgt[3]*fc[off[3]];
      s_fea[wave][lm][ca] = v;
    }
    __syncthreads();

    // Build A tiles (16x32 f16 each): row=lm; K map per ISA 7.12.2.
    v16h A0, A1;
    #pragma unroll
    for (int h = 0; h < 16; h++) {
      int K = (h < 8) ? (hs*8 + h) : (8 + hs*8 + h);
      A0[h] = (_Float16)s_fea[wave][lm][K];
      A1[h] = (_Float16)s_fea[wave][lm][32 + K];
    }
    // Compress: mid_all[16, 32] = fea[16,64] x WcT[64,32]
    v8f acc0 = {}, acc1 = {};
    acc0 = wmma_f16(A0, Bc0, acc0);   // (k 0-31,  n 0-15)
    acc0 = wmma_f16(A1, Bc2, acc0);   // (k 32-63, n 0-15)
    acc1 = wmma_f16(A0, Bc1, acc1);   // (k 0-31,  n 16-31)
    acc1 = wmma_f16(A1, Bc3, acc1);   // (k 32-63, n 16-31)

    // Stage raw per-expert mids: C/D layout -> LDS [pixel][n].
    #pragma unroll
    for (int j = 0; j < 8; j++) {
      int row = j + hs*8;
      s_mid[wave][row][lm]      = acc0[j];
      s_mid[wave][row][16 + lm] = acc1[j];
    }
    __syncthreads();

    // Routing mix: mid[p,o] = sum_e r_e * mid_all[p, e*8+o];
    // expand-A operand: A'[p, e*8+o] = r_e * mid[p,o] (A layout, K=n).
    float mixed[8];
    #pragma unroll
    for (int o = 0; o < 8; o++)
      mixed[o] = rt.x*s_mid[wave][lm][o]      + rt.y*s_mid[wave][lm][8 + o]
               + rt.z*s_mid[wave][lm][16 + o] + rt.w*s_mid[wave][lm][24 + o];
    v16h Am;
    #pragma unroll
    for (int h = 0; h < 16; h++) {
      float s = (h < 8) ? rlo : rhi;
      Am[h] = (_Float16)(s * mixed[h & 7]);
    }
    // Expand: out[16, 64] = A'[16,32] x WeT[32,64]
    v8f oc0 = {}, oc1 = {}, oc2 = {}, oc3 = {};
    oc0 = wmma_f16(Am, Be0, oc0);
    oc1 = wmma_f16(Am, Be1, oc1);
    oc2 = wmma_f16(Am, Be2, oc2);
    oc3 = wmma_f16(Am, Be3, oc3);

    // Epilogue: += fea0, transpose into [channel][pixel] for coalesced stores.
    #pragma unroll
    for (int j = 0; j < 8; j++) {
      int row = j + hs*8;
      s_out[wave][lm     ][row] = oc0[j] + s_fea[wave][row][lm];
      s_out[wave][16 + lm][row] = oc1[j] + s_fea[wave][row][16 + lm];
      s_out[wave][32 + lm][row] = oc2[j] + s_fea[wave][row][32 + lm];
      s_out[wave][48 + lm][row] = oc3[j] + s_fea[wave][row][48 + lm];
    }
    __syncthreads();

    // Store: 16 consecutive pixels per channel are contiguous (16 | WHR).
    int y = pb / WHR, xb = pb % WHR;
    float* outb = out + (size_t)b*CH*NPIX + (size_t)y*WHR + xb;
    #pragma unroll
    for (int it = 0; it < 8; it++) {
      int c  = it*8 + (lane >> 2);
      int p0 = (lane & 3) * 4;
      *(float4*)(outb + (size_t)c*NPIX + p0) = *(const float4*)&s_out[wave][c][p0];
    }
    __syncthreads();
  }
}

// ---------------------------------------------------------------------------
extern "C" void kernel_launch(void* const* d_in, const int* in_sizes, int n_in,
                              void* d_out, int out_size, void* d_ws, size_t ws_size,
                              hipStream_t stream) {
  (void)in_sizes; (void)n_in; (void)out_size; (void)ws_size;
  // setup_inputs order: x(0,unused), fused(1), wc(2), we(3), w1(4), b1(5),
  // w2(6), b2(7), rw(8), rb(9), ow(10), ob(11), scale(12,hardcoded=2)
  const float* fused = (const float*)d_in[1];
  const float* wc = (const float*)d_in[2];
  const float* we = (const float*)d_in[3];
  const float* w1 = (const float*)d_in[4];
  const float* b1 = (const float*)d_in[5];
  const float* w2 = (const float*)d_in[6];
  const float* b2 = (const float*)d_in[7];
  const float* rw = (const float*)d_in[8];
  const float* rb = (const float*)d_in[9];
  const float* ow = (const float*)d_in[10];
  const float* ob = (const float*)d_in[11];
  float* out = (float*)d_out;

  // Workspace carve (all 16B aligned):
  //  [0,       8192)                      : 8 packed f16 B-matrices (4096 halfs)
  //  [8192,    8192+NPIX*8)               : float2 sample coords
  //  [...,     +NPIX*16)                  : float4 routing weights
  char* ws = (char*)d_ws;
  _Float16* wsw   = (_Float16*)ws;
  float2* coords  = (float2*)(ws + 8192);
  float4* routg   = (float4*)(ws + 8192 + (size_t)NPIX*8);

  pack_weights_kernel<<<1, 32, 0, stream>>>(wc, we, wsw);
  mlp_kernel<<<NPIX/256, 256, 0, stream>>>(w1, b1, w2, b2, rw, rb, ow, ob,
                                           coords, routg);
  upsample_main_kernel<<<(NPIX/16)/4, 128, 0, stream>>>(fused, wsw, coords,
                                                        routg, out);
}
